// baselineModel_309237645993
// MI455X (gfx1250) — matrix operand order repaired and checked by reading, not compiled, not run on hardware
//
#include <hip/hip_runtime.h>
#include <hip/hip_bf16.h>

// Problem constants (match reference)
#define TT   256
#define BB   128
#define EE   256
#define HH   512
#define KK   7
#define G4H  2048   // 4*H

typedef __attribute__((ext_vector_type(16))) _Float16 v16h;
typedef __attribute__((ext_vector_type(8)))  float    v8f;
typedef __attribute__((ext_vector_type(4)))  unsigned int u32x4;
typedef __attribute__((ext_vector_type(8)))  unsigned int u32x8;

// ---------------------------------------------------------------------------
// CDNA5 async copy: global -> LDS DMA (ASYNCcnt tracked), 16B per lane.
// ---------------------------------------------------------------------------
__device__ __forceinline__ void async_copy_b128(const _Float16* gptr, _Float16* lptr) {
    unsigned lds            = (unsigned)(uintptr_t)lptr;       // flat low 32 = LDS offset
    unsigned long long addr = (unsigned long long)(uintptr_t)gptr;
    asm volatile("global_load_async_to_lds_b128 %0, %1, off"
                 :: "v"(lds), "v"(addr) : "memory");
}
#define WAIT_ASYNC_0() asm volatile("s_wait_asynccnt 0x0" ::: "memory")
#define WAIT_ASYNC_1() asm volatile("s_wait_asynccnt 0x1" ::: "memory")

// ---------------------------------------------------------------------------
// CDNA5 Tensor Data Mover: stage a 64-row x 32-f16 tile (row stride = Kdim
// elements) into LDS with a 16B pad after every 64B row -> 80B LDS rows.
// 2-D tensor => D# groups 0 and 1 only (VADDR2/3 NULL). TENSORcnt tracked.
// ---------------------------------------------------------------------------
__device__ __forceinline__ u32x8 tdm_make_group1(unsigned Kdim) {
    const unsigned td0 = 32u, td1 = 64u;          // tensor dims (== tile, all in-bounds)
    u32x8 g1;
    g1[0] = (1u << 16)        // data_size = 1 -> 2 bytes (f16)
          | (1u << 20)        // pad_enable
          | (3u << 22)        // pad_interval: 16 DWORDs stored per pad
          | (3u << 25);       // pad_amount : 4 DWORDs (16B)
    g1[1] = (td0 & 0xFFFFu) << 16;                        // [63:48] tensor_dim0 lo16
    g1[2] = (td0 >> 16) | ((td1 & 0xFFFFu) << 16);        // [79:64] td0 hi, [95:80] td1 lo
    g1[3] = (td1 >> 16) | (32u << 16);                    // [111:96] td1 hi, [127:112] tile_dim0 = 32
    g1[4] = 64u;                                          // [143:128] tile_dim1 = 64, tile_dim2 = 0
    g1[5] = Kdim;                                         // tensor_dim0_stride lo32 (elements)
    g1[6] = 0u;                                           // stride0 hi16 | stride1 lo16
    g1[7] = 0u;                                           // stride1 hi32
    return g1;
}

__device__ __forceinline__ void tdm_load_tile(const _Float16* gsrc, _Float16* ldst, u32x8 g1) {
    unsigned long long ga = (unsigned long long)(uintptr_t)gsrc;
    u32x4 g0;
    g0[0] = 1u;                                           // count=1, user descriptor
    g0[1] = (unsigned)(uintptr_t)ldst;                    // lds_addr
    g0[2] = (unsigned)ga;                                 // global_addr[31:0]
    g0[3] = (unsigned)((ga >> 32) & 0x01FFFFFFu) | (2u << 30);  // global_addr[56:32] | type=2
    asm volatile("tensor_load_to_lds %0, %1" :: "s"(g0), "s"(g1) : "memory");
}

// ---------------------------------------------------------------------------
// init: zero output scalar, cell state, and h f16 double buffers
// ---------------------------------------------------------------------------
__global__ void init_state_kernel(float* out, int nout, float* c, _Float16* h) {
    int i = blockIdx.x * blockDim.x + threadIdx.x;
    int stride = gridDim.x * blockDim.x;
    if (i < nout) out[i] = 0.f;
    const int nc = 2 * BB * HH;          // [dir][B][H] f32
    for (int j = i; j < nc; j += stride) c[j] = 0.f;
    const int nh = 2 * 2 * BB * HH;      // [dir][buf][B][H] f16
    for (int j = i; j < nh; j += stride) h[j] = (_Float16)0.f;
}

// ---------------------------------------------------------------------------
// f32 -> f16 weight conversion
// ---------------------------------------------------------------------------
__global__ void convert_f16_kernel(const float* __restrict__ src, _Float16* __restrict__ dst, int n) {
    int i = blockIdx.x * blockDim.x + threadIdx.x;
    int stride = gridDim.x * blockDim.x;
    for (; i < n; i += stride) dst[i] = (_Float16)src[i];
}

// bsum[0][:] = bih_f + bhh_f ; bsum[1][:] = bih_b + bhh_b
__global__ void bias_sum_kernel(const float* bif, const float* bhf,
                                const float* bib, const float* bhb, float* bsum) {
    int i = blockIdx.x * blockDim.x + threadIdx.x;
    if (i < G4H)            bsum[i] = bif[i] + bhf[i];
    else if (i < 2 * G4H) { int j = i - G4H; bsum[i] = bib[j] + bhb[j]; }
}

// ---------------------------------------------------------------------------
// embedding gather + f16 convert: xs_h[t*B+b][e] = table[idx[t][b]][e]
// ---------------------------------------------------------------------------
__global__ void embed_gather_kernel(const int* __restrict__ idx,
                                    const float* __restrict__ table,
                                    _Float16* __restrict__ xs) {
    int i = blockIdx.x * blockDim.x + threadIdx.x;            // TT*BB*EE threads
    int tb = i / EE;
    int e  = i % EE;
    int row = idx[tb];
    xs[i] = (_Float16)table[(size_t)row * EE + e];
}

// ---------------------------------------------------------------------------
// WMMA GEMM, C = A * B^T  (A: [M,K] f16 row-major, Bw: [N,K] f16 row-major,
// C: [M,ldc] f32). blockIdx.z selects direction via element strides.
// Block = 256 threads (8 waves), 64x64 output tile, K stepped by 32.
// Pipeline: A tiles via global_load_async_to_lds_b128 (ASYNCcnt), B tiles via
// TDM tensor_load_to_lds with LDS row padding (TENSORcnt); double-buffered.
// Each wave owns two 16x16 v_wmma_f32_16x16x32_f16 tiles.
// ---------------------------------------------------------------------------
__global__ void wmma_gemm_nt_kernel(const _Float16* __restrict__ A,
                                    const _Float16* __restrict__ Bw,
                                    float* __restrict__ C,
                                    int Kdim, int ldc,
                                    long long strideA, long long strideB, long long strideC) {
    A  += (long long)blockIdx.z * strideA;
    Bw += (long long)blockIdx.z * strideB;
    C  += (long long)blockIdx.z * strideC;

    __shared__ __align__(16) _Float16 lA[2][64][40];   // 64x32 tiles, rows padded to 80B
    __shared__ __align__(16) _Float16 lB[2][64][40];   // B^T tiles: lB[buf][n][k] (TDM pads rows)

    const int tid   = threadIdx.x;          // 0..255
    const int wave  = tid >> 5;             // 0..7
    const int lane  = tid & 31;
    const int half  = lane >> 4;            // 0/1
    const int mlane = lane & 15;
    const int mt    = wave & 3;             // M sub-tile 0..3
    const int nt0   = (wave >> 2) << 1;     // N sub-tiles {0,1} or {2,3}

    const int m0 = blockIdx.x * 64;
    const int n0 = blockIdx.y * 64;

    const int lr = tid >> 2;                // 0..63  (tile row to stage)
    const int lc = (tid & 3) << 3;          // 0,8,16,24 (halfword col)

    const size_t arow_g = (size_t)(m0 + lr) * Kdim + lc;
    const _Float16* bpanel = &Bw[(size_t)n0 * Kdim];   // 64-row weight panel

    const u32x8 g1 = tdm_make_group1((unsigned)Kdim);

    v8f acc0 = {};
    v8f acc1 = {};

    const int nk = Kdim >> 5;               // K chunks of 32

    // prologue: stage chunk 0 into buffer 0
    if (wave == 0) tdm_load_tile(bpanel, &lB[0][0][0], g1);
    async_copy_b128(&A[arow_g], &lA[0][lr][lc]);

    for (int kc = 0; kc < nk; ++kc) {
        const int cur = kc & 1;
        const int nxt = cur ^ 1;
        if (kc + 1 < nk) {
            const int k1 = (kc + 1) << 5;
            // buffers `nxt` were last read in iteration kc-1 (barrier-protected)
            if (wave == 0) tdm_load_tile(bpanel + k1, &lB[nxt][0][0], g1);
            async_copy_b128(&A[arow_g + k1], &lA[nxt][lr][lc]);
            __builtin_amdgcn_s_wait_tensorcnt((short)1);   // chunk kc's TDM landed
            WAIT_ASYNC_1();                                // chunk kc's A DMA landed
        } else {
            __builtin_amdgcn_s_wait_tensorcnt((short)0);
            WAIT_ASYNC_0();
        }
        __syncthreads();                    // all DMA for chunk kc visible block-wide

        // A fragment: lane (half h, row m): K = 16h .. 16h+15 (contiguous)
        union { v16h v; uint4 q[2]; } au, bu0, bu1;
        const int arow = mt * 16 + mlane;
        au.q[0] = *(const uint4*)&lA[cur][arow][16 * half];
        au.q[1] = *(const uint4*)&lA[cur][arow][16 * half + 8];

        // B fragment: lane n = mlane, K = 16*half .. 16*half+15 (contiguous)
        const int brow0 = nt0 * 16 + mlane;
        const int brow1 = (nt0 + 1) * 16 + mlane;
        bu0.q[0] = *(const uint4*)&lB[cur][brow0][16 * half];
        bu0.q[1] = *(const uint4*)&lB[cur][brow0][16 * half + 8];
        bu1.q[0] = *(const uint4*)&lB[cur][brow1][16 * half];
        bu1.q[1] = *(const uint4*)&lB[cur][brow1][16 * half + 8];

        acc0 = __builtin_amdgcn_wmma_f32_16x16x32_f16(false, au.v, false, bu0.v,
                                                      (short)0, acc0, false, false);
        acc1 = __builtin_amdgcn_wmma_f32_16x16x32_f16(false, au.v, false, bu1.v,
                                                      (short)0, acc1, false, false);
        __syncthreads();                    // protect `cur` before DMA reuses it
    }

    // C/D layout: VGPR r holds M = r + 8*half, N = mlane
    const int crow  = m0 + mt * 16 + 8 * half;
    const int ccol0 = n0 + nt0 * 16 + mlane;
    const int ccol1 = n0 + (nt0 + 1) * 16 + mlane;
#pragma unroll
    for (int r = 0; r < 8; ++r) {
        C[(size_t)(crow + r) * ldc + ccol0] = acc0[r];
        C[(size_t)(crow + r) * ldc + ccol1] = acc1[r];
    }
}

// ---------------------------------------------------------------------------
// LSTM pointwise step: gates = rec + gin[tf] + bias; update c, h.
// Gate order (PyTorch): i, f, g, o.  z = dir, tf = z? T-1-t : t.
// ---------------------------------------------------------------------------
__global__ void lstm_pointwise_kernel(const float* __restrict__ rec,   // [2][B][4H]
                                      const float* __restrict__ gin,   // [2][T*B][4H]
                                      const float* __restrict__ bsum,  // [2][4H]
                                      float* __restrict__ cst,         // [2][B][H]
                                      _Float16* __restrict__ hbuf,     // [2][2][B][H]
                                      float* __restrict__ hs,          // [T][B][2H]
                                      int t, int wbuf) {
    int i = blockIdx.x * blockDim.x + threadIdx.x;   // 2*B*H exact
    int z = i / (BB * HH);
    int r = i % (BB * HH);
    int b = r / HH;
    int j = r % HH;
    int tf = z ? (TT - 1 - t) : t;

    size_t gi = (size_t)z * TT * BB * G4H + (size_t)(tf * BB + b) * G4H;
    size_t ri = (size_t)z * BB * G4H + (size_t)b * G4H;
    const float* bz = bsum + z * G4H;

    float xi = rec[ri + j]            + gin[gi + j]            + bz[j];
    float xf = rec[ri + HH + j]       + gin[gi + HH + j]       + bz[HH + j];
    float xg = rec[ri + 2 * HH + j]   + gin[gi + 2 * HH + j]   + bz[2 * HH + j];
    float xo = rec[ri + 3 * HH + j]   + gin[gi + 3 * HH + j]   + bz[3 * HH + j];

    float si = 1.f / (1.f + __expf(-xi));
    float sf = 1.f / (1.f + __expf(-xf));
    float so = 1.f / (1.f + __expf(-xo));
    float tg = tanhf(xg);

    size_t ci = (size_t)z * BB * HH + (size_t)b * HH + j;
    float c = sf * cst[ci] + si * tg;
    float h = so * tanhf(c);
    cst[ci] = c;
    hbuf[(size_t)z * 2 * BB * HH + (size_t)wbuf * BB * HH + (size_t)b * HH + j] = (_Float16)h;
    hs[(size_t)(tf * BB + b) * 2 * HH + (size_t)z * HH + j] = h;
}

// ---------------------------------------------------------------------------
// emissions: em[b][t][k] = hs[t][b][:] . lin_W[k][:] + lin_b[k]
// lin_W staged in LDS (7*1024*4 = 28 KB). One thread per (t,b).
// ---------------------------------------------------------------------------
__global__ void emissions_kernel(const float* __restrict__ hs,
                                 const float* __restrict__ linW,
                                 const float* __restrict__ linb,
                                 float* __restrict__ em) {
    __shared__ float sW[KK][2 * HH];
    for (int i = threadIdx.x; i < KK * 2 * HH; i += blockDim.x)
        sW[i / (2 * HH)][i % (2 * HH)] = linW[i];
    __syncthreads();

    int tb = blockIdx.x * blockDim.x + threadIdx.x;   // exactly T*B
    int t = tb / BB, b = tb % BB;
    float acc[KK];
#pragma unroll
    for (int k = 0; k < KK; ++k) acc[k] = linb[k];
    const float* hrow = hs + (size_t)tb * 2 * HH;
    for (int j = 0; j < 2 * HH; ++j) {
        float h = hrow[j];
#pragma unroll
        for (int k = 0; k < KK; ++k) acc[k] += h * sW[k][j];
    }
#pragma unroll
    for (int k = 0; k < KK; ++k)
        em[(size_t)b * TT * KK + (size_t)t * KK + k] = acc[k];
}

// ---------------------------------------------------------------------------
// CRF NLL: one wave per batch row. alpha recursion across lanes 0..6 via shfl,
// gold-path score on lane 0, atomicAdd of (logZ - score) into the scalar out.
// ---------------------------------------------------------------------------
__global__ void crf_kernel(const float* __restrict__ em,      // [B][T][K]
                           const int* __restrict__ labels,    // [B][T]
                           const float* __restrict__ trans,   // [K][K]
                           const float* __restrict__ strans,  // [K]
                           const float* __restrict__ etrans,  // [K]
                           float* __restrict__ out) {
    const int b = blockIdx.x;
    const int lane = threadIdx.x;     // 32
    __shared__ float tr[KK * KK];
    for (int i = lane; i < KK * KK; i += 32) tr[i] = trans[i];
    __syncthreads();

    const float* eb = em + (size_t)b * TT * KK;
    const int*   lb = labels + (size_t)b * TT;
    const int ln = (lane < KK) ? lane : 0;

    float alpha = (lane < KK) ? (strans[lane] + eb[lane]) : -1e30f;
    for (int t = 1; t < TT; ++t) {
        float prev[KK];
#pragma unroll
        for (int k = 0; k < KK; ++k) prev[k] = __shfl(alpha, k, 32);
        float mx = -1e30f;
#pragma unroll
        for (int k = 0; k < KK; ++k) mx = fmaxf(mx, prev[k] + tr[k * KK + ln]);
        float s = 0.f;
#pragma unroll
        for (int k = 0; k < KK; ++k) s += __expf(prev[k] + tr[k * KK + ln] - mx);
        float na = mx + __logf(s) + eb[t * KK + ln];
        alpha = (lane < KK) ? na : -1e30f;
    }
    float v = (lane < KK) ? (alpha + etrans[lane]) : -1e30f;
    float mx = -1e30f;
#pragma unroll
    for (int k = 0; k < KK; ++k) mx = fmaxf(mx, __shfl(v, k, 32));
    float s = 0.f;
#pragma unroll
    for (int k = 0; k < KK; ++k) s += __expf(__shfl(v, k, 32) - mx);
    float logZ = mx + __logf(s);

    if (lane == 0) {
        float score = strans[lb[0]] + etrans[lb[TT - 1]];
        for (int t = 0; t < TT; ++t) score += eb[t * KK + lb[t]];
        for (int t = 0; t < TT - 1; ++t) score += tr[lb[t] * KK + lb[t + 1]];
        atomicAdd(out, logZ - score);   // accumulate -(score - logZ)
    }
}

// ---------------------------------------------------------------------------
// launch
// ---------------------------------------------------------------------------
extern "C" void kernel_launch(void* const* d_in, const int* in_sizes, int n_in,
                              void* d_out, int out_size, void* d_ws, size_t ws_size,
                              hipStream_t stream) {
    (void)in_sizes; (void)n_in; (void)ws_size;

    const int*   inputData = (const int*)d_in[0];
    const int*   labels    = (const int*)d_in[1];
    const float* table     = (const float*)d_in[2];
    const float* Wih_f     = (const float*)d_in[3];
    const float* Whh_f     = (const float*)d_in[4];
    const float* bih_f     = (const float*)d_in[5];
    const float* bhh_f     = (const float*)d_in[6];
    const float* Wih_b     = (const float*)d_in[7];
    const float* Whh_b     = (const float*)d_in[8];
    const float* bih_b     = (const float*)d_in[9];
    const float* bhh_b     = (const float*)d_in[10];
    const float* lin_W     = (const float*)d_in[11];
    const float* lin_b     = (const float*)d_in[12];
    const float* trans     = (const float*)d_in[13];
    const float* strans    = (const float*)d_in[14];
    const float* etrans    = (const float*)d_in[15];
    float* out = (float*)d_out;

    // workspace carve-up (256B aligned)
    char* ws = (char*)d_ws;
    size_t off = 0;
    auto take = [&](size_t bytes) { size_t o = off; off += (bytes + 255) & ~(size_t)255; return o; };
    _Float16* xs_h  = (_Float16*)(ws + take((size_t)TT * BB * EE * 2));       // [T*B][E]
    _Float16* wih_h = (_Float16*)(ws + take((size_t)2 * G4H * EE * 2));       // [2][4H][E]
    _Float16* whh_h = (_Float16*)(ws + take((size_t)2 * G4H * HH * 2));       // [2][4H][H]
    float*    bsum  = (float*)   (ws + take((size_t)2 * G4H * 4));            // [2][4H]
    float*    gin   = (float*)   (ws + take((size_t)2 * TT * BB * G4H * 4));  // [2][T*B][4H]
    float*    grec  = (float*)   (ws + take((size_t)2 * BB * G4H * 4));       // [2][B][4H]
    _Float16* hbuf  = (_Float16*)(ws + take((size_t)2 * 2 * BB * HH * 2));    // [2][2][B][H]
    float*    cbuf  = (float*)   (ws + take((size_t)2 * BB * HH * 4));        // [2][B][H]
    float*    hs    = (float*)   (ws + take((size_t)TT * BB * 2 * HH * 4));   // [T][B][2H]
    float*    em    = (float*)   (ws + take((size_t)BB * TT * KK * 4));       // [B][T][K]

    // 0) init state + output
    init_state_kernel<<<1024, 256, 0, stream>>>(out, out_size, cbuf, hbuf);

    // 1) weight / bias prep
    convert_f16_kernel<<<512, 256, 0, stream>>>(Wih_f, wih_h,               G4H * EE);
    convert_f16_kernel<<<512, 256, 0, stream>>>(Wih_b, wih_h + G4H * EE,    G4H * EE);
    convert_f16_kernel<<<512, 256, 0, stream>>>(Whh_f, whh_h,               G4H * HH);
    convert_f16_kernel<<<512, 256, 0, stream>>>(Whh_b, whh_h + G4H * HH,    G4H * HH);
    bias_sum_kernel<<<16, 256, 0, stream>>>(bih_f, bhh_f, bih_b, bhh_b, bsum);

    // 2) embedding gather -> f16
    embed_gather_kernel<<<(TT * BB * EE) / 256, 256, 0, stream>>>(inputData, table, xs_h);

    // 3) input projection for all timesteps & both dirs:
    //    gin[z] = xs @ Wih[z]^T   (M=T*B=32768, N=4H=2048, K=E=256)
    wmma_gemm_nt_kernel<<<dim3(TT * BB / 64, G4H / 64, 2), 256, 0, stream>>>(
        xs_h, wih_h, gin, EE, G4H,
        /*strideA=*/0, /*strideB=*/(long long)G4H * EE, /*strideC=*/(long long)TT * BB * G4H);

    // 4) recurrence: 256 steps, both dirs concurrently (blockIdx.z)
    for (int t = 0; t < TT; ++t) {
        const int rb = t & 1;         // read buffer
        const int wb = (t + 1) & 1;   // write buffer
        // grec[z] = h_prev[z] @ Whh[z]^T  (M=128, N=2048, K=512)
        wmma_gemm_nt_kernel<<<dim3(BB / 64, G4H / 64, 2), 256, 0, stream>>>(
            hbuf + (size_t)rb * BB * HH, whh_h, grec, HH, G4H,
            /*strideA=*/(long long)2 * BB * HH, /*strideB=*/(long long)G4H * HH,
            /*strideC=*/(long long)BB * G4H);
        lstm_pointwise_kernel<<<(2 * BB * HH) / 256, 256, 0, stream>>>(
            grec, gin, bsum, cbuf, hbuf, hs, t, wb);
    }

    // 5) emissions
    emissions_kernel<<<(TT * BB) / 256, 256, 0, stream>>>(hs, lin_W, lin_b, em);

    // 6) CRF NLL (accumulates into zeroed out[0])
    crf_kernel<<<BB, 32, 0, stream>>>(em, labels, trans, strans, etrans, out);
}
